// E60cForgetGateCell_45792941310473
// MI455X (gfx1250) — matrix operand order repaired and checked by reading, not compiled, not run on hardware
//
#include <hip/hip_runtime.h>
#include <hip/hip_bf16.h>

// ---------------------------------------------------------------------------
// Forget-gate RNN cell for MI455X (gfx1250, wave32, WMMA bf16 16x16x32).
//   Phase 1: convert x + 4 weight matrices to bf16 (scratch).
//   Phase 2: fused GEMM: Wx_all = x @ W_x^T, Wfx_all = x @ W_f^T (f32 accum).
//   Phase 3: 1024 sequential step kernels; each does h@W_h^T and h@U_f^T with
//            v_wmma_f32_16x16x32_bf16, fused gate epilogue, bf16 h ping-pong.
// ---------------------------------------------------------------------------

typedef __attribute__((ext_vector_type(16))) __bf16 v16bf;
typedef __attribute__((ext_vector_type(8)))  __bf16 v8bf;
typedef __attribute__((ext_vector_type(8)))  float  v8f;

static constexpr int T_STEPS = 1024;
static constexpr int BATCH   = 32;
static constexpr int DIM     = 1024;

__device__ __forceinline__ float fast_sigmoid(float x) {
    return 1.0f / (1.0f + __expf(-x));   // x=+inf -> 1, x=-inf -> 0 (safe)
}
__device__ __forceinline__ float fast_tanh(float x) {
    // 1 - 2/(e^{2x}+1): saturates to +1 when e==inf, -1 when e==0 (NaN-free).
    float e = __expf(2.0f * x);
    return 1.0f - 2.0f / (e + 1.0f);
}

// ---- WMMA tile loaders (row-major sources, contiguous 128-bit loads) -------
// A tile: 16x32 bf16, act row-major [M_total, DIM].
// lanes 0-15  (row m_base+lane): K = [k0..k0+7]   and [k0+16..k0+23]
// lanes 16-31 (row m_base+l-16): K = [k0+8..k0+15] and [k0+24..k0+31]
__device__ __forceinline__ v16bf load_a_tile(const __bf16* A, int m_base,
                                             int k0, int lane) {
    const int lh = lane & 15;
    const int hi = lane >> 4;
    const __bf16* p = A + (size_t)(m_base + lh) * DIM + k0 + hi * 8;
    v16bf a;
    *((v8bf*)&a)     = *(const v8bf*)p;         // global_load_b128
    *((v8bf*)&a + 1) = *(const v8bf*)(p + 16);  // global_load_b128
    return a;
}

// B tile: 32x16 bf16 from weight row-major [N=DIM, K=DIM] (out = act @ W^T).
// lanes 0-15  (col n_base+lane): K = [k0..k0+7]   and [k0+16..k0+23]
// lanes 16-31 (col n_base+l-16): K = [k0+8..k0+15] and [k0+24..k0+31]
__device__ __forceinline__ v16bf load_b_tile(const __bf16* W, int n_base,
                                             int k0, int lane) {
    const int lh = lane & 15;
    const int hi = lane >> 4;
    const __bf16* p = W + (size_t)(n_base + lh) * DIM + k0 + hi * 8;
    v16bf b;
    *((v8bf*)&b)     = *(const v8bf*)p;
    *((v8bf*)&b + 1) = *(const v8bf*)(p + 16);
    return b;
}

__device__ __forceinline__ v8f wmma_bf16(v16bf a, v16bf b, v8f c) {
    return __builtin_amdgcn_wmma_f32_16x16x32_bf16(
        /*neg_a=*/false, a, /*neg_b=*/false, b,
        /*c_mod=*/(short)0, c, /*reuse_a=*/false, /*reuse_b=*/false);
}

// ---- Phase 1: f32 -> bf16 conversion ---------------------------------------
__global__ void cvt_f32_bf16_kernel(const float* __restrict__ src,
                                    __bf16* __restrict__ dst, int n) {
    int i = blockIdx.x * blockDim.x + threadIdx.x;
    int stride = gridDim.x * blockDim.x;
    for (; i < n; i += stride) dst[i] = (__bf16)src[i];
}

// h0 -> hseq row 0 (f32) and hbf0 (bf16)
__global__ void init_h_kernel(const float* __restrict__ h0,
                              float* __restrict__ hseq,
                              __bf16* __restrict__ hbf) {
    int i = blockIdx.x * blockDim.x + threadIdx.x;
    if (i < BATCH * DIM) {
        float v = h0[i];
        hseq[i] = v;
        hbf[i]  = (__bf16)v;
    }
}

// ---- Phase 2: fused input projections (both gates share the A tile) --------
// grid = (T*B/32, DIM/64), block = 256 (8 waves: 2 M-tiles x 4 N-tiles)
__global__ void xproj_kernel(const __bf16* __restrict__ xbf,
                             const __bf16* __restrict__ Wxbf,
                             const __bf16* __restrict__ Wfbf,
                             float* __restrict__ wx_all,
                             float* __restrict__ wfx_all) {
    const int lane = threadIdx.x & 31;
    const int wave = threadIdx.x >> 5;
    const int m_base = blockIdx.x * 32 + (wave & 1) * 16;
    const int n_base = blockIdx.y * 64 + (wave >> 1) * 16;

    v8f cx = {};
    v8f cf = {};
    for (int k0 = 0; k0 < DIM; k0 += 32) {
        v16bf a  = load_a_tile(xbf,  m_base, k0, lane);
        v16bf bx = load_b_tile(Wxbf, n_base, k0, lane);
        v16bf bf = load_b_tile(Wfbf, n_base, k0, lane);
        cx = wmma_bf16(a, bx, cx);
        cf = wmma_bf16(a, bf, cf);
    }

    const int n  = n_base + (lane & 15);
    const int mo = (lane >> 4) * 8;
#pragma unroll
    for (int r = 0; r < 8; ++r) {
        size_t idx = (size_t)(m_base + mo + r) * DIM + n;
        wx_all[idx]  = cx[r];
        wfx_all[idx] = cf[r];
    }
}

// ---- Phase 3: one recurrent step -------------------------------------------
// grid = (BATCH/16, DIM/16), block = 32 (one wave per 16x16 output tile,
// accumulating both gates). hbf double-buffered across launches.
__global__ void step_kernel(const __bf16* __restrict__ hbf_in,
                            __bf16* __restrict__ hbf_out,
                            const __bf16* __restrict__ Whbf,
                            const __bf16* __restrict__ Ufbf,
                            const float* __restrict__ wx_all,
                            const float* __restrict__ wfx_all,
                            const float* __restrict__ b,
                            const float* __restrict__ b_f,
                            float* __restrict__ outs,
                            float* __restrict__ hseq,
                            int t) {
    const int lane = threadIdx.x & 31;
    const int m_base = blockIdx.x * 16;
    const int n_base = blockIdx.y * 16;

    v8f cu = {};   // h @ W_h^T
    v8f cf = {};   // h @ U_f^T
    for (int k0 = 0; k0 < DIM; k0 += 32) {
        v16bf a  = load_a_tile(hbf_in, m_base, k0, lane);
        v16bf bu = load_b_tile(Whbf,   n_base, k0, lane);
        v16bf bf = load_b_tile(Ufbf,   n_base, k0, lane);
        cu = wmma_bf16(a, bu, cu);
        cf = wmma_bf16(a, bf, cf);
    }

    const int n   = n_base + (lane & 15);
    const int mo  = (lane >> 4) * 8;
    const float bn  = b[n];
    const float bfn = b_f[n];
    const size_t tslab = (size_t)t * BATCH * DIM;

#pragma unroll
    for (int r = 0; r < 8; ++r) {
        const int m = m_base + mo + r;
        const size_t idx  = (size_t)m * DIM + n;
        const size_t tidx = tslab + idx;

        float forget = fast_sigmoid(wfx_all[tidx] + cf[r] + bfn);
        float update = fast_tanh  (wx_all[tidx]  + cu[r] + bn);
        float hp     = hseq[tidx];                    // h row t
        float hn     = forget * hp + (1.0f - forget) * update;

        hseq[tidx + (size_t)BATCH * DIM] = hn;        // h row t+1
        outs[tidx]    = hn * hn * fast_sigmoid(hn);   // h * silu(h)
        hbf_out[idx]  = (__bf16)hn;                   // bf16 state for next A
    }
}

// ---------------------------------------------------------------------------
extern "C" void kernel_launch(void* const* d_in, const int* in_sizes, int n_in,
                              void* d_out, int out_size, void* d_ws,
                              size_t ws_size, hipStream_t stream) {
    (void)in_sizes; (void)n_in; (void)out_size; (void)ws_size;

    // setup_inputs order: x, h0, W_h, W_x, W_f, U_f, b, b_f
    const float* x   = (const float*)d_in[0];
    const float* h0  = (const float*)d_in[1];
    const float* W_h = (const float*)d_in[2];
    const float* W_x = (const float*)d_in[3];
    const float* W_f = (const float*)d_in[4];
    const float* U_f = (const float*)d_in[5];
    const float* b   = (const float*)d_in[6];
    const float* b_f = (const float*)d_in[7];

    // d_out = outs [T,B,D] ++ h [T+1,B,D], all f32
    float* outs = (float*)d_out;
    float* hseq = (float*)d_out + (size_t)T_STEPS * BATCH * DIM;

    // workspace layout
    const size_t SZ_XBF  = (size_t)T_STEPS * BATCH * DIM * sizeof(__bf16);
    const size_t SZ_WBF  = (size_t)DIM * DIM * sizeof(__bf16);
    const size_t SZ_PROJ = (size_t)T_STEPS * BATCH * DIM * sizeof(float);
    const size_t SZ_HBF  = (size_t)BATCH * DIM * sizeof(__bf16);

    char* p = (char*)d_ws;
    __bf16* xbf   = (__bf16*)p; p += SZ_XBF;
    __bf16* Whbf  = (__bf16*)p; p += SZ_WBF;
    __bf16* Wxbf  = (__bf16*)p; p += SZ_WBF;
    __bf16* Wfbf  = (__bf16*)p; p += SZ_WBF;
    __bf16* Ufbf  = (__bf16*)p; p += SZ_WBF;
    float* wx_all  = (float*)p; p += SZ_PROJ;
    float* wfx_all = (float*)p; p += SZ_PROJ;
    __bf16* hbf0  = (__bf16*)p; p += SZ_HBF;
    __bf16* hbf1  = (__bf16*)p; p += SZ_HBF;

    // Phase 1: conversions
    const int nX = T_STEPS * BATCH * DIM;
    const int nW = DIM * DIM;
    cvt_f32_bf16_kernel<<<4096, 256, 0, stream>>>(x,   xbf,  nX);
    cvt_f32_bf16_kernel<<<1024, 256, 0, stream>>>(W_h, Whbf, nW);
    cvt_f32_bf16_kernel<<<1024, 256, 0, stream>>>(W_x, Wxbf, nW);
    cvt_f32_bf16_kernel<<<1024, 256, 0, stream>>>(W_f, Wfbf, nW);
    cvt_f32_bf16_kernel<<<1024, 256, 0, stream>>>(U_f, Ufbf, nW);
    init_h_kernel<<<(BATCH * DIM + 255) / 256, 256, 0, stream>>>(h0, hseq, hbf0);

    // Phase 2: batched input projections (both gates fused)
    dim3 pgrid((T_STEPS * BATCH) / 32, DIM / 64);
    xproj_kernel<<<pgrid, 256, 0, stream>>>(xbf, Wxbf, Wfbf, wx_all, wfx_all);

    // Phase 3: sequential scan, bf16 h state ping-pong
    __bf16* hb[2] = {hbf0, hbf1};
    dim3 sgrid(BATCH / 16, DIM / 16);
    for (int t = 0; t < T_STEPS; ++t) {
        step_kernel<<<sgrid, 32, 0, stream>>>(
            hb[t & 1], hb[(t + 1) & 1], Whbf, Ufbf,
            wx_all, wfx_all, b, b_f, outs, hseq, t);
    }
}
